// RoGCNLayer_4733053960666
// MI455X (gfx1250) — matrix operand order repaired and checked by reading, not compiled, not run on hardware
//
#include <hip/hip_runtime.h>

typedef __attribute__((ext_vector_type(16))) _Float16     v16h;
typedef __attribute__((ext_vector_type(8)))  float        v8f;
typedef __attribute__((ext_vector_type(4)))  unsigned int u32x4;
typedef __attribute__((ext_vector_type(8)))  int          i32x8;
typedef __attribute__((ext_vector_type(4)))  int          i32x4;

#define N_ENT  500000
#define N_REL  64
#define D_IN   128
#define N_EDGE 8000000

// swizzled B panel: frag f = kb*8+nt (16 frags) x lane (32) x half (16) = 8192 f16 = 16 KB
#define SWZB_ELEMS (16 * 32 * 16)

// ---------------------------------------------------------------- zero counts
__global__ void __launch_bounds__(256)
zero_counts_kernel(u32x4* counts4, int n4) {
    int i = blockIdx.x * blockDim.x + threadIdx.x;
    if (i < n4) counts4[i] = (u32x4){0u, 0u, 0u, 0u};
}

// ---------------------------------------------------------------- edge scatter
// counts (128 MB) fits in the 192 MB L2: non-returning GLOBAL_ATOMIC_ADD_U32
// resolves in the L2 atomic units. Edge lists are read-once -> NT loads.
__global__ void __launch_bounds__(256)
scatter_edges_kernel(const int* __restrict__ src,
                     const int* __restrict__ etype,
                     unsigned int* __restrict__ counts, int nedge) {
    int i = blockIdx.x * blockDim.x + threadIdx.x;
    if (i < nedge) {
        int s = __builtin_nontemporal_load(src + i);
        int r = __builtin_nontemporal_load(etype + i);
        atomicAdd(&counts[(size_t)s * N_REL + r], 1u);
    }
}

// ---------------------------------------------------------------- B swizzle
// Pack rel_emb (f32 [64][128]) into per-lane WMMA B-fragment order so the GEMM
// kernel reads each fragment as 32 contiguous bytes (2x ds_load_b128).
//   swzB[((kb*8+nt)*32 + lane)*16 + i] = f16(rel_emb[kb*32 + (i>>3)*16 + (lane>>4)*8 + (i&7)][nt*16 + (lane&15)])
__global__ void __launch_bounds__(256)
build_swzb_kernel(const float* __restrict__ rel_emb, _Float16* __restrict__ swzB) {
    int idx = blockIdx.x * blockDim.x + threadIdx.x;
    if (idx < SWZB_ELEMS) {
        int i    = idx & 15;
        int lane = (idx >> 4) & 31;
        int f    = idx >> 9;             // 0..15
        int kb   = f >> 3, nt = f & 7;
        int k = kb * 32 + (i >> 3) * 16 + (lane >> 4) * 8 + (i & 7);
        int n = nt * 16 + (lane & 15);
        swzB[idx] = (_Float16)rel_emb[k * D_IN + n];
    }
}

// ---------------------------------------------------------------- WMMA GEMM + epilogue
// 256 threads = 8 waves; block = 128 rows, wave = 16-row slab x 128 cols.
// B panel staged LDS-side by the Tensor Data Mover (TENSORcnt).
__global__ void __launch_bounds__(256)
rgcn_wmma_kernel(const unsigned int* __restrict__ counts,
                 const _Float16* __restrict__ swzB,
                 const float* __restrict__ bias,
                 float* __restrict__ out) {
    __shared__ __align__(32) _Float16 lds_b[SWZB_ELEMS];   // 16 KB of 320 KB WGP LDS

    const int tid  = threadIdx.x;
    const int wave = tid >> 5;
    const int lane = tid & 31;

    // ---- TDM: DMA the 16 KB swizzled B panel global -> LDS (wave 0 issues).
    if (wave == 0) {
        unsigned           lds_addr = (unsigned)(size_t)(&lds_b[0]);
        unsigned long long ga       = (unsigned long long)(size_t)swzB;
        // D# group 0: count=1 (valid user descriptor), lds_addr, global_addr, type=2
        u32x4 g0;
        g0[0] = 1u;
        g0[1] = lds_addr;
        g0[2] = (unsigned)(ga & 0xffffffffu);
        g0[3] = (unsigned)((ga >> 32) & 0x01ffffffu) | 0x80000000u;  // type=2 in bits 127:126
        // D# group 1: data_size=8B, 1-D tile of 2048 units (16 KB)
        i32x8 g1;
        g1[0] = 0x00030000;            // data_size=3 (8 bytes); wg_mask=0; no flags
        g1[1] = 0x08000000;            // tensor_dim0[15:0]=2048 (bits 63:48)
        g1[2] = 0x00010000;            // tensor_dim1=1 (bits 95:80)
        g1[3] = 0x08000000;            // tile_dim0=2048 (bits 127:112)
        g1[4] = 0x00000001;            // tile_dim1=1, tile_dim2=0
        g1[5] = 2048;                  // tensor_dim0_stride (low 32)
        g1[6] = 0x08000000;            // tensor_dim1_stride[15:0]=2048 (bits 223:208)
        g1[7] = 0;
        i32x4 gz4 = {0, 0, 0, 0};
        i32x8 gz8 = {0, 0, 0, 0, 0, 0, 0, 0};
        __builtin_amdgcn_tensor_load_to_lds(g0, g1, gz4, gz4, gz8, 0);  // 6-arg form
        __builtin_amdgcn_s_wait_tensorcnt(0);               // TENSORcnt == 0
    }
    __syncthreads();

    const int row0 = blockIdx.x * 128 + wave * 16;
    const int m    = lane & 15;        // A layout: lane = M
    const int hi   = lane >> 4;
    const int kg   = hi * 8;           // A: K-subgroup per lane half

    // Clamp reads (EXEC must stay all-ones for WMMA); guard stores instead.
    int row  = row0 + m;
    int rowc = row < N_ENT ? row : (N_ENT - 1);
    const unsigned int* crow = counts + (size_t)rowc * N_REL;

    // ---- A fragments (counts as f16, exact for small ints) + degree.
    v16h a[2];
    unsigned int degsum = 0u;
    #pragma unroll
    for (int kb = 0; kb < 2; ++kb) {
        const int kbase = kb * 32;
        u32x4 q0 = __builtin_nontemporal_load((const u32x4*)(crow + kbase + kg));
        u32x4 q1 = __builtin_nontemporal_load((const u32x4*)(crow + kbase + kg + 4));
        u32x4 q2 = __builtin_nontemporal_load((const u32x4*)(crow + kbase + 16 + kg));
        u32x4 q3 = __builtin_nontemporal_load((const u32x4*)(crow + kbase + 16 + kg + 4));
        unsigned int v[16] = { q0[0],q0[1],q0[2],q0[3], q1[0],q1[1],q1[2],q1[3],
                               q2[0],q2[1],q2[2],q2[3], q3[0],q3[1],q3[2],q3[3] };
        v16h af;
        #pragma unroll
        for (int i = 0; i < 16; ++i) {
            af[i] = (_Float16)(float)v[i];
            degsum += v[i];
        }
        a[kb] = af;
    }
    // Row m's counts split between lanes m and m+16 -> xor-16 reduce.
    float deg  = (float)degsum + (float)__shfl_xor((int)degsum, 16, 32);
    float invd = 1.0f / (deg + 1.0f);

    // ---- 16x V_WMMA_F32_16X16X32_F16: 8 col tiles x {K 0-31, K 32-63}
    const int n = lane & 15;           // B/C layout: lane = N
    v8f c[8];
    #pragma unroll
    for (int nt = 0; nt < 8; ++nt) c[nt] = (v8f){0,0,0,0,0,0,0,0};

    #pragma unroll
    for (int nt = 0; nt < 8; ++nt) {
        #pragma unroll
        for (int kb = 0; kb < 2; ++kb) {
            const v16h bf = *(const v16h*)&lds_b[((kb * 8 + nt) * 32 + lane) * 16];
            c[nt] = __builtin_amdgcn_wmma_f32_16x16x32_f16(
                        false, a[kb], false, bf, (short)0, c[nt], false, false);
        }
    }

    // ---- epilogue: relu(c * invd[row] + bias[col]); output is streaming -> NT stores.
    float bv[8];
    #pragma unroll
    for (int nt = 0; nt < 8; ++nt) bv[nt] = bias[nt * 16 + n];

    #pragma unroll
    for (int r = 0; r < 8; ++r) {
        const int   mrow = hi * 8 + r;             // C layout: VGPR r -> row r / r+8
        const float id   = __shfl(invd, mrow, 32);
        const int   orow = row0 + mrow;
        if (orow < N_ENT) {
            float* op = out + (size_t)orow * D_IN;
            #pragma unroll
            for (int nt = 0; nt < 8; ++nt) {
                float v = c[nt][r] * id + bv[nt];
                v = v > 0.0f ? v : 0.0f;
                __builtin_nontemporal_store(v, &op[nt * 16 + n]);
            }
        }
    }
}

// ---------------------------------------------------------------- launch
extern "C" void kernel_launch(void* const* d_in, const int* in_sizes, int n_in,
                              void* d_out, int out_size, void* d_ws, size_t ws_size,
                              hipStream_t stream) {
    // setup_inputs order: num_entities(scalar), source, edge_type, rel_emb, bias
    const int*   source = (const int*)d_in[1];
    const int*   etype  = (const int*)d_in[2];
    const float* rel    = (const float*)d_in[3];
    const float* bias   = (const float*)d_in[4];
    float*       out    = (float*)d_out;

    // ws: counts[N_ENT][N_REL] u32 (128 MB, L2-resident) | swizzled B (16 KB)
    unsigned int* counts = (unsigned int*)d_ws;
    _Float16*     swzB   = (_Float16*)((char*)d_ws + (size_t)N_ENT * N_REL * sizeof(unsigned int));

    const int n4 = (N_ENT * N_REL) / 4;
    zero_counts_kernel<<<(n4 + 255) / 256, 256, 0, stream>>>((u32x4*)counts, n4);

    scatter_edges_kernel<<<(N_EDGE + 255) / 256, 256, 0, stream>>>(
        source, etype, counts, N_EDGE);

    build_swzb_kernel<<<(SWZB_ELEMS + 255) / 256, 256, 0, stream>>>(rel, swzB);

    const int nblocks = (N_ENT + 127) / 128;
    rgcn_wmma_kernel<<<nblocks, 256, 0, stream>>>(counts, swzB, bias, out);
}
